// PPO_Craftax_Agent_26010321945192
// MI455X (gfx1250) — matrix operand order repaired and checked by reading, not compiled, not run on hardware
//
#include <hip/hip_runtime.h>
#include <cstdint>
#include <cstddef>

typedef __attribute__((ext_vector_type(16))) __bf16 v16bf;
typedef __attribute__((ext_vector_type(8)))  float  v8f;

constexpr int kT = 128;   // timesteps
constexpr int kB = 256;   // batch
constexpr int kH = 128;   // hidden
constexpr int kG = 512;   // 4*H gates
constexpr int kOBS = 8268;
constexpr int kWIDTH = 1024;
constexpr int kENC = 512;
constexpr int kM = kT * kB;  // 32768 rows

constexpr int BM = 128, BN = 128, BK = 32;
constexpr int LDSK = 40;   // padded LDS K-stride (20 banks -> conflict-free b128 frags)

__device__ __forceinline__ unsigned short f32_to_bf16(float f) {
  unsigned int u = __builtin_bit_cast(unsigned int, f);
  u += 0x7FFFu + ((u >> 16) & 1u);           // round-to-nearest-even
  return (unsigned short)(u >> 16);
}

// tanh(x) = 1 - 2/(exp(2x)+1); saturates correctly: exp->inf => 1, exp->0 => -1.
// Uses v_exp_f32 (trans) + v_rcp_f32 instead of the IEEE divide expansion.
__device__ __forceinline__ float fast_tanh(float x) {
  float e = __expf(2.0f * x);
  return 1.0f - 2.0f * __builtin_amdgcn_rcpf(e + 1.0f);
}

__device__ __forceinline__ float fast_sigmoid(float x) {
  return __builtin_amdgcn_rcpf(1.0f + __expf(-x));
}

union Pack16 { unsigned short s[16]; uint2 d[4]; uint4 q[2]; };
union Frag   { uint4 q[2]; v16bf v; };
union F16    { float4 f4[4]; float f[16]; };

// ---------------- conversion kernels ----------------
__global__ void cvt_f32_to_bf16_kernel(const float* __restrict__ in,
                                       unsigned short* __restrict__ out, size_t n) {
  size_t i = (size_t)blockIdx.x * blockDim.x + threadIdx.x;
  if (i < n) out[i] = f32_to_bf16(in[i]);
}

__global__ void add_bias_kernel(const float* __restrict__ a, const float* __restrict__ b,
                                float* __restrict__ out, int n) {
  int i = blockIdx.x * blockDim.x + threadIdx.x;
  if (i < n) out[i] = a[i] + b[i];
}

// ---------------- fused GEMM: C = act(A[M,K] * Bw[N,K]^T + bias[N]) ----------------
// A: f32 (converted in-flight) or bf16; Bw: bf16 row-major [N,K]; C: bf16 or f32.
template <bool A_F32, bool ACT_TANH, bool OUT_F32>
__global__ __launch_bounds__(256) void gemm_bt_kernel(
    const void* __restrict__ Aptr, const unsigned short* __restrict__ Bw,
    const float* __restrict__ bias, void* __restrict__ Cptr,
    int M, int N, int K) {
  __shared__ unsigned short As[BM * LDSK];
  __shared__ unsigned short Bs[BN * LDSK];

  const int tid = threadIdx.x;
  const int m0 = blockIdx.y * BM;
  const int n0 = blockIdx.x * BN;

  const int lrow = tid >> 1;      // 0..127: tile row being staged
  const int half = tid & 1;       // which 16-element K half

  const int wid  = tid >> 5;      // 8 waves
  const int lane = tid & 31;
  const int wm = wid & 3;         // 4 waves along M (32 rows each)
  const int wn = wid >> 2;        // 2 waves along N (64 cols each)
  const int l15 = lane & 15;
  const int hi  = lane >> 4;      // half-wave select

  const v8f vzero = {0.f, 0.f, 0.f, 0.f, 0.f, 0.f, 0.f, 0.f};
  v8f acc[2][4];
#pragma unroll
  for (int ms = 0; ms < 2; ++ms)
#pragma unroll
    for (int ns = 0; ns < 4; ++ns) acc[ms][ns] = vzero;

  for (int k0 = 0; k0 < K; k0 += BK) {
    const int kbase = k0 + half * 16;
    Pack16 pa, pb;
    // ---- stage A chunk into registers (convert f32->bf16 if needed) ----
    if (A_F32) {
      const float* Af = (const float*)Aptr + (size_t)(m0 + lrow) * K;
      F16 uf;
      if (kbase + 16 <= K) {
        const float4* s = (const float4*)(Af + kbase);
#pragma unroll
        for (int j = 0; j < 4; ++j) uf.f4[j] = s[j];
      } else {
#pragma unroll
        for (int j = 0; j < 16; ++j) uf.f[j] = (kbase + j < K) ? Af[kbase + j] : 0.f;
      }
#pragma unroll
      for (int j = 0; j < 16; ++j) pa.s[j] = f32_to_bf16(uf.f[j]);
    } else {
      const unsigned short* Ab = (const unsigned short*)Aptr + (size_t)(m0 + lrow) * K;
      if (kbase + 16 <= K) {
        const uint2* s = (const uint2*)(Ab + kbase);
#pragma unroll
        for (int j = 0; j < 4; ++j) pa.d[j] = s[j];
      } else {
#pragma unroll
        for (int j = 0; j < 16; ++j)
          pa.s[j] = (kbase + j < K) ? Ab[kbase + j] : (unsigned short)0;
      }
    }
    // ---- stage B (weight) chunk ----
    {
      const unsigned short* Bb = Bw + (size_t)(n0 + lrow) * K;
      if (kbase + 16 <= K) {
        const uint2* s = (const uint2*)(Bb + kbase);
#pragma unroll
        for (int j = 0; j < 4; ++j) pb.d[j] = s[j];
      } else {
#pragma unroll
        for (int j = 0; j < 16; ++j)
          pb.s[j] = (kbase + j < K) ? Bb[kbase + j] : (unsigned short)0;
      }
    }
    // prefetch next K-tile (global_prefetch_b8)
    if (k0 + BK < K) {
      if (A_F32)
        __builtin_prefetch((const float*)Aptr + (size_t)(m0 + lrow) * K + kbase + BK, 0, 1);
      else
        __builtin_prefetch((const unsigned short*)Aptr + (size_t)(m0 + lrow) * K + kbase + BK, 0, 1);
      __builtin_prefetch(Bw + (size_t)(n0 + lrow) * K + kbase + BK, 0, 1);
    }

    __syncthreads();   // previous iteration's fragment reads complete
    {
      uint4* da = (uint4*)&As[lrow * LDSK + half * 16];
      da[0] = pa.q[0]; da[1] = pa.q[1];
      uint4* db = (uint4*)&Bs[lrow * LDSK + half * 16];
      db[0] = pb.q[0]; db[1] = pb.q[1];
    }
    __syncthreads();

    // ---- fragments per ISA layout ----
    // A 16x32 bf16: lanes0-15 K{0..7,16..23}, lanes16-31 K{8..15,24..31}
    Frag af[2];
#pragma unroll
    for (int ms = 0; ms < 2; ++ms) {
      const unsigned short* p = &As[(wm * 32 + ms * 16 + l15) * LDSK + hi * 8];
      af[ms].q[0] = *(const uint4*)p;
      af[ms].q[1] = *(const uint4*)(p + 16);
    }
    // B 32x16 bf16: lanes0-15 K{0..7,16..23} (col=lane), lanes16-31 K{8..15,24..31}
    Frag bfr[4];
#pragma unroll
    for (int ns = 0; ns < 4; ++ns) {
      const unsigned short* p = &Bs[(wn * 64 + ns * 16 + l15) * LDSK + hi * 8];
      bfr[ns].q[0] = *(const uint4*)p;
      bfr[ns].q[1] = *(const uint4*)(p + 16);
    }
#pragma unroll
    for (int ms = 0; ms < 2; ++ms)
#pragma unroll
      for (int ns = 0; ns < 4; ++ns)
        acc[ms][ns] = __builtin_amdgcn_wmma_f32_16x16x32_bf16(
            false, af[ms].v, false, bfr[ns].v, (short)0, acc[ms][ns], false, false);
  }

  // ---- epilogue: bias + optional tanh, C layout VGPR j -> row j (+8 for hi half) ----
#pragma unroll
  for (int ns = 0; ns < 4; ++ns) {
    const int col = n0 + wn * 64 + ns * 16 + l15;
    const float bv = bias[col];
#pragma unroll
    for (int ms = 0; ms < 2; ++ms) {
      const int rbase = m0 + wm * 32 + ms * 16 + hi * 8;
#pragma unroll
      for (int j = 0; j < 8; ++j) {
        float v = acc[ms][ns][j] + bv;
        if (ACT_TANH) v = fast_tanh(v);
        const size_t off = (size_t)(rbase + j) * N + col;
        if (OUT_F32) ((float*)Cptr)[off] = v;
        else ((unsigned short*)Cptr)[off] = f32_to_bf16(v);
      }
    }
  }
}

// ---------------- LSTM: batch-parallel recurrence, WMMA h@Whh^T per step ----------------
constexpr int LDSK2 = 136;  // padded K-stride for the 16x128 A tile

__global__ __launch_bounds__(256) void lstm_kernel(
    const float* __restrict__ xg,     // [T*B, 4H] precomputed x-gates (+biases)
    const float* __restrict__ done,   // [T*B]
    const float* __restrict__ h0, const float* __restrict__ c0,   // [B,H]
    const unsigned short* __restrict__ Whhbf,                      // [4H,H] bf16
    unsigned short* __restrict__ hsbf) {                           // [T*B,H] bf16 out
  __shared__ float hC[16 * kH];
  __shared__ float cC[16 * kH];
  __shared__ unsigned short hA[16 * LDSK2];
  __shared__ float gates[16 * kG];

  const int tid = threadIdx.x;
  const int rb = blockIdx.x * 16;    // 16 batch rows per block
  const int wid = tid >> 5;
  const int lane = tid & 31;
  const int l15 = lane & 15;
  const int hi = lane >> 4;
  const int colb = wid * 64;         // 8 waves x 64 gate columns

  for (int i = tid; i < 16 * kH; i += 256) {
    int r = i >> 7, k = i & (kH - 1);
    hC[i] = h0[(rb + r) * kH + k];
    cC[i] = c0[(rb + r) * kH + k];
  }

  for (int t = 0; t < kT; ++t) {
    __syncthreads();
    // mask by done, pack masked h to bf16 A-tile
    for (int i = tid; i < 16 * kH; i += 256) {
      int r = i >> 7, k = i & (kH - 1);
      float m = 1.0f - done[t * kB + rb + r];
      float hv = hC[i] * m;
      cC[i] = cC[i] * m;
      hA[r * LDSK2 + k] = f32_to_bf16(hv);
    }
    __syncthreads();

    // gates[16,512] = xg[t] + h @ Whh^T   (M=16, N=512, K=128)
    v8f acc[4];
#pragma unroll
    for (int ns = 0; ns < 4; ++ns) {
      const int col = colb + ns * 16 + l15;
      const size_t rowb = (size_t)(t * kB + rb + hi * 8);
#pragma unroll
      for (int j = 0; j < 8; ++j) acc[ns][j] = xg[(rowb + j) * kG + col];
    }
    Frag af[4];
#pragma unroll
    for (int kf = 0; kf < 4; ++kf) {
      const unsigned short* p = &hA[l15 * LDSK2 + kf * 32 + hi * 8];
      af[kf].q[0] = *(const uint4*)p;
      af[kf].q[1] = *(const uint4*)(p + 16);
    }
#pragma unroll
    for (int ns = 0; ns < 4; ++ns) {
      const int n = colb + ns * 16 + l15;
#pragma unroll
      for (int kf = 0; kf < 4; ++kf) {
        const unsigned short* p = &Whhbf[n * kH + kf * 32 + hi * 8];
        Frag bfr;
        bfr.q[0] = *(const uint4*)p;
        bfr.q[1] = *(const uint4*)(p + 16);
        acc[ns] = __builtin_amdgcn_wmma_f32_16x16x32_bf16(
            false, af[kf].v, false, bfr.v, (short)0, acc[ns], false, false);
      }
    }
#pragma unroll
    for (int ns = 0; ns < 4; ++ns) {
      const int col = colb + ns * 16 + l15;
#pragma unroll
      for (int j = 0; j < 8; ++j) gates[(hi * 8 + j) * kG + col] = acc[ns][j];
    }
    __syncthreads();

    // cell update: gate order i,f,g,o
    for (int i = tid; i < 16 * kH; i += 256) {
      int r = i >> 7, k = i & (kH - 1);
      float gi = gates[r * kG + k];
      float gf = gates[r * kG + kH + k];
      float gg = gates[r * kG + 2 * kH + k];
      float go = gates[r * kG + 3 * kH + k];
      float si = fast_sigmoid(gi);
      float sf = fast_sigmoid(gf);
      float so = fast_sigmoid(go);
      float c = sf * cC[i] + si * fast_tanh(gg);
      cC[i] = c;
      float h = so * fast_tanh(c);
      hC[i] = h;
      hsbf[(size_t)(t * kB + rb + r) * kH + k] = f32_to_bf16(h);
    }
  }
}

// ---------------- launch ----------------
extern "C" void kernel_launch(void* const* d_in, const int* in_sizes, int n_in,
                              void* d_out, int out_size, void* d_ws, size_t ws_size,
                              hipStream_t stream) {
  (void)in_sizes; (void)n_in; (void)out_size; (void)ws_size;
  const float* x    = (const float*)d_in[0];
  const float* done = (const float*)d_in[1];
  const float* h0   = (const float*)d_in[2];
  const float* c0   = (const float*)d_in[3];
  const float* W0   = (const float*)d_in[4];
  const float* b0   = (const float*)d_in[5];
  const float* W1   = (const float*)d_in[6];
  const float* b1   = (const float*)d_in[7];
  const float* W2   = (const float*)d_in[8];
  const float* b2   = (const float*)d_in[9];
  const float* Wo   = (const float*)d_in[10];
  const float* bo   = (const float*)d_in[11];
  const float* Wih  = (const float*)d_in[12];
  const float* Whh  = (const float*)d_in[13];
  const float* bih  = (const float*)d_in[14];
  const float* bhh  = (const float*)d_in[15];
  const float* Wp   = (const float*)d_in[16];
  const float* bp   = (const float*)d_in[17];
  float* out = (float*)d_out;

  char* p = (char*)d_ws;
  auto take = [&](size_t bytes) -> char* {
    char* r = p;
    p += (bytes + 255) & ~(size_t)255;
    return r;
  };
  unsigned short* W0bf  = (unsigned short*)take((size_t)kWIDTH * kOBS * 2);
  unsigned short* W1bf  = (unsigned short*)take((size_t)kWIDTH * kWIDTH * 2);
  unsigned short* W2bf  = (unsigned short*)take((size_t)kWIDTH * kWIDTH * 2);
  unsigned short* Wobf  = (unsigned short*)take((size_t)kENC * kWIDTH * 2);
  unsigned short* Wihbf = (unsigned short*)take((size_t)kG * kENC * 2);
  unsigned short* Whhbf = (unsigned short*)take((size_t)kG * kH * 2);
  unsigned short* Wpbf  = (unsigned short*)take((size_t)kH * kH * 2);
  float*          biasIH= (float*)take((size_t)kG * 4);
  unsigned short* buf0  = (unsigned short*)take((size_t)kM * kWIDTH * 2);
  unsigned short* buf1  = (unsigned short*)take((size_t)kM * kWIDTH * 2);
  unsigned short* encbf = (unsigned short*)take((size_t)kM * kENC * 2);
  float*          xg    = (float*)take((size_t)kM * kG * 4);
  unsigned short* hsbf  = (unsigned short*)take((size_t)kM * kH * 2);

  auto cvt = [&](const float* in, unsigned short* o, size_t n) {
    int blocks = (int)((n + 255) / 256);
    cvt_f32_to_bf16_kernel<<<blocks, 256, 0, stream>>>(in, o, n);
  };
  cvt(W0, W0bf, (size_t)kWIDTH * kOBS);
  cvt(W1, W1bf, (size_t)kWIDTH * kWIDTH);
  cvt(W2, W2bf, (size_t)kWIDTH * kWIDTH);
  cvt(Wo, Wobf, (size_t)kENC * kWIDTH);
  cvt(Wih, Wihbf, (size_t)kG * kENC);
  cvt(Whh, Whhbf, (size_t)kG * kH);
  cvt(Wp, Wpbf, (size_t)kH * kH);
  add_bias_kernel<<<2, 256, 0, stream>>>(bih, bhh, biasIH, kG);

  dim3 blk(256);
  // encoder MLP (tanh after every layer)
  gemm_bt_kernel<true,  true,  false><<<dim3(kWIDTH / BN, kM / BM), blk, 0, stream>>>(
      x,     W0bf, b0, buf0,  kM, kWIDTH, kOBS);
  gemm_bt_kernel<false, true,  false><<<dim3(kWIDTH / BN, kM / BM), blk, 0, stream>>>(
      buf0,  W1bf, b1, buf1,  kM, kWIDTH, kWIDTH);
  gemm_bt_kernel<false, true,  false><<<dim3(kWIDTH / BN, kM / BM), blk, 0, stream>>>(
      buf1,  W2bf, b2, buf0,  kM, kWIDTH, kWIDTH);
  gemm_bt_kernel<false, true,  false><<<dim3(kENC / BN, kM / BM), blk, 0, stream>>>(
      buf0,  Wobf, bo, encbf, kM, kENC,   kWIDTH);
  // hoisted gate projection: xg = enc @ Wih^T + (bih+bhh), f32, no activation
  gemm_bt_kernel<false, false, true ><<<dim3(kG / BN, kM / BM), blk, 0, stream>>>(
      encbf, Wihbf, biasIH, xg, kM, kG, kENC);
  // recurrence (batch-parallel, no cross-block sync needed)
  lstm_kernel<<<dim3(kB / 16), blk, 0, stream>>>(xg, done, h0, c0, Whhbf, hsbf);
  // post_lstm head -> f32 output
  gemm_bt_kernel<false, true,  true ><<<dim3(kH / BN, kM / BM), blk, 0, stream>>>(
      hsbf, Wpbf, bp, out, kM, kH, kH);
}
